// InMambaEncoder_34127810134213
// MI455X (gfx1250) — compile-verified, matrix operand order checked by reading, not run on hardware
//
#include <hip/hip_runtime.h>
#include <hip/hip_bf16.h>

// ---------------- problem constants ----------------
#define BB   4
#define LL   512
#define BLT  2048          // B*L
#define DM   512
#define DFF  512
#define DIN  256
#define NS   16            // D_STATE
#define RR   32            // DT_RANK
#define ELAYERS 2

typedef __attribute__((ext_vector_type(16))) __bf16 v16bf_t;
typedef __attribute__((ext_vector_type(8)))  float  v8f_t;

__device__ __forceinline__ float silu_f(float x)     { return x / (1.f + __expf(-x)); }
__device__ __forceinline__ float softplus_f(float x) { return x > 20.f ? x : log1pf(__expf(x)); }

__device__ __forceinline__ void pack8(v16bf_t& v, int base, float4 a, float4 b)
{
    v[base + 0] = (__bf16)a.x; v[base + 1] = (__bf16)a.y;
    v[base + 2] = (__bf16)a.z; v[base + 3] = (__bf16)a.w;
    v[base + 4] = (__bf16)b.x; v[base + 5] = (__bf16)b.y;
    v[base + 6] = (__bf16)b.z; v[base + 7] = (__bf16)b.w;
}

// ---------------------------------------------------------------------------
// Batched GEMM: C[b] = act(A[b] (MxK) @ B[b] (KxN, or NxK if TRB) + bias)
// One wave computes a 16x(16*NT) tile via v_wmma_f32_16x16x32_bf16.
// Requirements (all calls satisfy them): M%16==0, N%(16*NT)==0, K%32 in {0,16}.
// No lane-divergent control flow: EXEC is all-ones around every WMMA.
// ACT: 0=none, 2=softplus
// ---------------------------------------------------------------------------
template<int ACT, int TRB, int NT>
__global__ __launch_bounds__(32)
void gemm_k(const float* __restrict__ A, long lda, long sA,
            const float* __restrict__ Bw, long ldb, long sB,
            const float* __restrict__ bias,
            float* __restrict__ C, long ldc, long sC, int K)
{
    const int lane = threadIdx.x;          // wave32
    const int half = lane >> 4;            // 0/1
    const int lr   = lane & 15;
    const int tm   = blockIdx.y << 4;
    const int tn0  = blockIdx.x * (16 * NT);

    const float* Ab = A  + (long)blockIdx.z * sA;
    const float* Bb = Bw + (long)blockIdx.z * sB;
    float*       Cb = C  + (long)blockIdx.z * sC;

    const int kA = half << 3;              // per-lane K sub-offset (ISA A layout)
    const int m  = tm + lr;
    const float* Arow = Ab + (long)m * lda;

    v8f_t acc[NT];
#pragma unroll
    for (int t = 0; t < NT; ++t) acc[t] = (v8f_t){};

    int k0 = 0;
    for (; k0 + 32 <= K; k0 += 32) {
        v16bf_t af;
        {
            const float4* p = (const float4*)(Arow + k0 + kA);
            const float4* q = (const float4*)(Arow + k0 + 16 + kA);
            pack8(af, 0, p[0], p[1]);
            pack8(af, 8, q[0], q[1]);
        }
#pragma unroll
        for (int t = 0; t < NT; ++t) {
            const int n = tn0 + t * 16 + lr;
            v16bf_t bfv;
            if (TRB) {
                const float* Brow = Bb + (long)n * ldb;
                const float4* p = (const float4*)(Brow + k0 + kA);
                const float4* q = (const float4*)(Brow + k0 + 16 + kA);
                pack8(bfv, 0, p[0], p[1]);
                pack8(bfv, 8, q[0], q[1]);
            } else {
#pragma unroll
                for (int i = 0; i < 8; ++i) {
                    bfv[i]     = (__bf16)Bb[(long)(k0 + kA + i) * ldb + n];
                    bfv[i + 8] = (__bf16)Bb[(long)(k0 + 16 + kA + i) * ldb + n];
                }
            }
            acc[t] = __builtin_amdgcn_wmma_f32_16x16x32_bf16(
                         false, af, false, bfv, (short)0, acc[t], false, false);
        }
    }
    if (k0 < K) {                           // uniform remainder of exactly 16
        v16bf_t af;
        {
            const float4* p = (const float4*)(Arow + k0 + kA);
            pack8(af, 0, p[0], p[1]);
#pragma unroll
            for (int i = 0; i < 8; ++i) af[i + 8] = (__bf16)0.f;
        }
#pragma unroll
        for (int t = 0; t < NT; ++t) {
            const int n = tn0 + t * 16 + lr;
            v16bf_t bfv;
            if (TRB) {
                const float* Brow = Bb + (long)n * ldb;
                const float4* p = (const float4*)(Brow + k0 + kA);
                pack8(bfv, 0, p[0], p[1]);
            } else {
#pragma unroll
                for (int i = 0; i < 8; ++i)
                    bfv[i] = (__bf16)Bb[(long)(k0 + kA + i) * ldb + n];
            }
#pragma unroll
            for (int i = 0; i < 8; ++i) bfv[i + 8] = (__bf16)0.f;
            acc[t] = __builtin_amdgcn_wmma_f32_16x16x32_bf16(
                         false, af, false, bfv, (short)0, acc[t], false, false);
        }
    }

    const int mb = half << 3;               // D layout: M = vgpr index (+8 upper half)
#pragma unroll
    for (int t = 0; t < NT; ++t) {
        const int n = tn0 + t * 16 + lr;
        const float bv = bias ? bias[n] : 0.f;
#pragma unroll
        for (int r = 0; r < 8; ++r) {
            float v = acc[t][r] + bv;
            if (ACT == 2) v = softplus_f(v);
            Cb[(long)(tm + mb + r) * ldc + n] = v;
        }
    }
}

// ---------------------------------------------------------------------------
// Selective-scan: per (batch, channel) 16-state linear recurrence over L.
// A_log is (1,16) broadcast over channels. Bm chunk staged in LDS.
// ---------------------------------------------------------------------------
__global__ void scan_kernel(const float* __restrict__ delta,   // (BL, d)
                            const float* __restrict__ u,       // (BL, d)
                            const float* __restrict__ Bm,      // (BL, *) ld=ldB
                            long ldB,
                            const float* __restrict__ Alog,    // (16)
                            float* __restrict__ hsum,          // (BL, d)
                            int d)
{
    __shared__ float sB[64 * NS];
    const int b  = blockIdx.x;
    const int ch = threadIdx.x;

    float An[NS];
#pragma unroll
    for (int n = 0; n < NS; ++n) An[n] = -__expf(Alog[n]);
    float h[NS];
#pragma unroll
    for (int n = 0; n < NS; ++n) h[n] = 0.f;

    for (int t0 = 0; t0 < LL; t0 += 64) {
        __syncthreads();
        for (int idx = threadIdx.x; idx < 64 * NS; idx += blockDim.x) {
            int tt = idx >> 4, n = idx & 15;
            sB[idx] = Bm[((long)b * LL + t0 + tt) * ldB + n];
        }
        __syncthreads();
        for (int tt = 0; tt < 64; ++tt) {
            long row = (long)b * LL + t0 + tt;
            float dl = delta[row * d + ch];
            float du = dl * u[row * d + ch];
            float s = 0.f;
#pragma unroll
            for (int n = 0; n < NS; ++n) {
                float a = __expf(dl * An[n]);
                h[n] = fmaf(a, h[n], du * sB[tt * NS + n]);
                s += h[n];
            }
            hsum[row * d + ch] = s;
        }
    }
}

// ---------------------------------------------------------------------------
// Row softmax over 512 columns with pre-scale (1/D_STATE). One block per row.
// ---------------------------------------------------------------------------
__global__ void softmax512(float* __restrict__ S, float scale)
{
    __shared__ float red[256];
    float* p = S + (long)blockIdx.x * 512;
    const int t = threadIdx.x;
    float v0 = p[t] * scale, v1 = p[t + 256] * scale;
    red[t] = fmaxf(v0, v1);
    __syncthreads();
    for (int s = 128; s > 0; s >>= 1) {
        if (t < s) red[t] = fmaxf(red[t], red[t + s]);
        __syncthreads();
    }
    float m = red[0];
    __syncthreads();
    float e0 = __expf(v0 - m), e1 = __expf(v1 - m);
    red[t] = e0 + e1;
    __syncthreads();
    for (int s = 128; s > 0; s >>= 1) {
        if (t < s) red[t] += red[t + s];
        __syncthreads();
    }
    float inv = 1.f / red[0];
    p[t]       = e0 * inv;
    p[t + 256] = e1 * inv;
}

// ---------------------------------------------------------------------------
// Elementwise helpers
// ---------------------------------------------------------------------------
__global__ void make_xz(const float* __restrict__ hs0, float* __restrict__ xx,
                        float* __restrict__ z, float* __restrict__ xf, float* __restrict__ zf)
{
    long i = (long)blockIdx.x * blockDim.x + threadIdx.x;
    if (i >= (long)BLT * DFF) return;
    int  c   = (int)(i & (DFF - 1));
    long row = i >> 9;
    int  b   = (int)(row >> 9);
    int  t   = (int)(row & (LL - 1));
    float vx = silu_f(hs0[row * 1024 + c]);
    float vz = silu_f(hs0[row * 1024 + DFF + c]);
    xx[i] = vx;
    z[i]  = vz;
    long fr = ((long)b * LL + (LL - 1 - t)) * DFF + c;
    xf[fr] = silu_f(vx);
    zf[fr] = silu_f(vz);
}

__global__ void flip_l(const float* __restrict__ in, float* __restrict__ out, int d)
{
    long i = (long)blockIdx.x * blockDim.x + threadIdx.x;
    if (i >= (long)BLT * d) return;
    int  c   = (int)(i % d);
    long row = i / d;
    int  b   = (int)(row / LL);
    int  t   = (int)(row % LL);
    out[((long)b * LL + (LL - 1 - t)) * d + c] = in[i];
}

__global__ void add_Du(float* __restrict__ y, long ldy, const float* __restrict__ Dm,
                       long ldd, const float* __restrict__ u, int d)
{
    long i = (long)blockIdx.x * blockDim.x + threadIdx.x;
    if (i >= (long)BLT * d) return;
    int  c   = (int)(i % d);
    long row = i / d;
    y[row * ldy + c] += Dm[row * ldd + c] * u[i];
}

__global__ void cat_add(const float* __restrict__ hs0, const float* __restrict__ oi,
                        const float* __restrict__ up, float* __restrict__ cat2)
{
    long i = (long)blockIdx.x * blockDim.x + threadIdx.x;
    if (i >= (long)BLT * 1024) return;
    int  j   = (int)(i & 1023);
    long row = i >> 10;
    float v = hs0[i];
    v += (j < DFF) ? oi[row * DFF + j] : up[row * DFF + (j - DFF)];
    cat2[i] = v;
}

__global__ void silu_out(const float* __restrict__ x, float* __restrict__ o, long n)
{
    long i = (long)blockIdx.x * blockDim.x + threadIdx.x;
    if (i < n) o[i] = silu_f(x[i]);
}

// ---------------------------------------------------------------------------
// Host orchestration
// ---------------------------------------------------------------------------
extern "C" void kernel_launch(void* const* d_in, const int* in_sizes, int n_in,
                              void* d_out, int out_size, void* d_ws, size_t ws_size,
                              hipStream_t stream)
{
    (void)in_sizes; (void)n_in; (void)out_size; (void)ws_size;

    const float* x_in   = (const float*)d_in[0];
    const float* in_w   = (const float*)d_in[1];
    const float* in_b   = (const float*)d_in[2];
    const float* xpw    = (const float*)d_in[3];
    const float* dtw    = (const float*)d_in[4];
    const float* dtb    = (const float*)d_in[5];
    const float* Alog   = (const float*)d_in[6];
    const float* exw    = (const float*)d_in[7];
    const float* exb    = (const float*)d_in[8];
    const float* ezw    = (const float*)d_in[9];
    const float* ezb    = (const float*)d_in[10];
    const float* xpew   = (const float*)d_in[11];
    const float* dtew   = (const float*)d_in[12];
    const float* dteb   = (const float*)d_in[13];
    const float* Aloge  = (const float*)d_in[14];
    const float* oiw    = (const float*)d_in[15];
    const float* oib    = (const float*)d_in[16];
    const float* oew    = (const float*)d_in[17];
    const float* oeb    = (const float*)d_in[18];
    const float* upw    = (const float*)d_in[19];
    const float* upb    = (const float*)d_in[20];
    const float* outw   = (const float*)d_in[21];
    const float* outb   = (const float*)d_in[22];
    const float* qiw    = (const float*)d_in[23];
    const float* qib    = (const float*)d_in[24];
    const float* kiw    = (const float*)d_in[25];
    const float* kib    = (const float*)d_in[26];
    const float* qew    = (const float*)d_in[27];
    const float* qeb    = (const float*)d_in[28];
    const float* kew    = (const float*)d_in[29];
    const float* keb    = (const float*)d_in[30];

    // ---- workspace carve-out (fp32) ----
    float* wsp = (float*)d_ws;
    size_t off = 0;
    auto alloc = [&](size_t n) { float* p = wsp + off; off += n; return p; };
    float* xbuf      = alloc((size_t)BLT * DM);
    float* hs0       = alloc((size_t)BLT * 1024);
    float* xx        = alloc((size_t)BLT * DFF);
    float* zb        = alloc((size_t)BLT * DFF);
    float* xf        = alloc((size_t)BLT * DFF);
    float* zf        = alloc((size_t)BLT * DFF);
    float* ub1       = alloc((size_t)BLT * DIN);
    float* ub2       = alloc((size_t)BLT * DIN);
    float* dbcd      = alloc((size_t)BLT * (DFF + 2 * NS + RR));
    float* delta     = alloc((size_t)BLT * DFF);
    float* hsum      = alloc((size_t)BLT * DFF);
    float* qbuf      = alloc((size_t)BLT * NS);
    float* kbuf      = alloc((size_t)BLT * NS);
    float* scores    = alloc((size_t)BB * LL * LL);
    float* ycat_in   = alloc((size_t)BLT * 1024);
    float* ycat_ex   = alloc((size_t)BLT * 512);
    float* out_inner = alloc((size_t)BLT * DFF);
    float* out_exit  = alloc((size_t)BLT * DIN);
    float* flipb     = alloc((size_t)BLT * DIN);
    float* upbuf     = alloc((size_t)BLT * DFF);
    float* cat2      = alloc((size_t)BLT * 1024);

    // dispatch helper: picks template instantiation
    auto gemm = [&](const float* A, long lda, long sA,
                    const float* Bw, long ldb, long sB,
                    const float* bias, float* C, long ldc, long sC,
                    int M, int N, int K, int nb, int tB, int act) {
        const int nt = (N % 32 == 0) ? 2 : 1;
        dim3 blk(32);
        dim3 g((unsigned)(N / (16 * nt)), (unsigned)(M / 16), (unsigned)nb);
        if (tB) {
            if (nt == 2) gemm_k<0, 1, 2><<<g, blk, 0, stream>>>(A, lda, sA, Bw, ldb, sB, bias, C, ldc, sC, K);
            else         gemm_k<0, 1, 1><<<g, blk, 0, stream>>>(A, lda, sA, Bw, ldb, sB, bias, C, ldc, sC, K);
        } else if (act == 2) {
            if (nt == 2) gemm_k<2, 0, 2><<<g, blk, 0, stream>>>(A, lda, sA, Bw, ldb, sB, bias, C, ldc, sC, K);
            else         gemm_k<2, 0, 1><<<g, blk, 0, stream>>>(A, lda, sA, Bw, ldb, sB, bias, C, ldc, sC, K);
        } else {
            if (nt == 2) gemm_k<0, 0, 2><<<g, blk, 0, stream>>>(A, lda, sA, Bw, ldb, sB, bias, C, ldc, sC, K);
            else         gemm_k<0, 0, 1><<<g, blk, 0, stream>>>(A, lda, sA, Bw, ldb, sB, bias, C, ldc, sC, K);
        }
    };

    auto run_ssm = [&](const float* u, int d,
                       const float* xw, const float* dw, const float* db,
                       const float* Al, const float* qw, const float* qb,
                       const float* kw, const float* kb, bool useD,
                       float* ydst, long ldy) {
        const long ldx = d + 2 * NS + RR;                 // dbcd leading dim
        gemm(u, d, 0, xw, ldx, 0, nullptr, dbcd, ldx, 0, BLT, (int)ldx, d, 1, 0, 0);
        gemm(dbcd, ldx, 0, dw, d, 0, db, delta, d, 0, BLT, d, RR, 1, 0, 2);
        scan_kernel<<<dim3(BB), dim3((unsigned)d), 0, stream>>>(
            delta, u, dbcd + RR, ldx, Al, hsum, d);
        gemm(u, d, 0, qw, NS, 0, qb, qbuf, NS, 0, BLT, NS, d, 1, 0, 2);
        gemm(dbcd + RR + NS, ldx, 0, kw, NS, 0, kb, kbuf, NS, 0, BLT, NS, NS, 1, 0, 2);
        gemm(qbuf, NS, (long)LL * NS, kbuf, NS, (long)LL * NS, nullptr,
             scores, LL, (long)LL * LL, LL, LL, NS, BB, 1, 0);
        softmax512<<<dim3(BLT), dim3(256), 0, stream>>>(scores, 1.f / (float)NS);
        gemm(scores, LL, (long)LL * LL, hsum, d, (long)LL * d, nullptr,
             ydst, ldy, (long)LL * ldy, LL, d, LL, BB, 0, 0);
        if (useD) {
            long total = (long)BLT * d;
            add_Du<<<dim3((unsigned)((total + 255) / 256)), dim3(256), 0, stream>>>(
                ydst, ldy, dbcd + RR + 2 * NS, ldx, u, d);
        }
    };

    const long nEW = (long)BLT * DFF;
    const dim3 eb(256);

    for (int l = 0; l < ELAYERS; ++l) {
        const float* xin = (l == 0) ? x_in : xbuf;
        const float* Lin_w  = in_w  + (long)l * DM * 1024;
        const float* Lin_b  = in_b  + (long)l * 1024;
        const float* Lxpw   = xpw   + (long)l * DFF * (DFF + 64);
        const float* Ldtw   = dtw   + (long)l * RR * DFF;
        const float* Ldtb   = dtb   + (long)l * DFF;
        const float* LAlog  = Alog  + (long)l * NS;
        const float* Lexw   = exw   + (long)l * DFF * DIN;
        const float* Lexb   = exb   + (long)l * DIN;
        const float* Lezw   = ezw   + (long)l * DFF * DIN;
        const float* Lezb   = ezb   + (long)l * DIN;
        const float* Lxpew  = xpew  + (long)l * DIN * (DIN + 64);
        const float* Ldtew  = dtew  + (long)l * RR * DIN;
        const float* Ldteb  = dteb  + (long)l * DIN;
        const float* LAloge = Aloge + (long)l * NS;
        const float* Loiw   = oiw   + (long)l * 1024 * DFF;
        const float* Loib   = oib   + (long)l * DFF;
        const float* Loew   = oew   + (long)l * 512 * DIN;
        const float* Loeb   = oeb   + (long)l * DIN;
        const float* Lupw   = upw   + (long)l * DIN * DFF;
        const float* Lupb   = upb   + (long)l * DFF;
        const float* Loutw  = outw  + (long)l * 1024 * DM;
        const float* Loutb  = outb  + (long)l * DM;
        const float* Lqiw   = qiw   + (long)l * DFF * NS;
        const float* Lqib   = qib   + (long)l * NS;
        const float* Lkiw   = kiw   + (long)l * NS * NS;
        const float* Lkib   = kib   + (long)l * NS;
        const float* Lqew   = qew   + (long)l * DIN * NS;
        const float* Lqeb   = qeb   + (long)l * NS;
        const float* Lkew   = kew   + (long)l * NS * NS;
        const float* Lkeb   = keb   + (long)l * NS;

        gemm(xin, DM, 0, Lin_w, 1024, 0, Lin_b, hs0, 1024, 0, BLT, 1024, DM, 1, 0, 0);
        make_xz<<<dim3((unsigned)((nEW + 255) / 256)), eb, 0, stream>>>(hs0, xx, zb, xf, zf);
        gemm(xf, DFF, 0, Lexw, DIN, 0, Lexb, ub1, DIN, 0, BLT, DIN, DFF, 1, 0, 0);
        gemm(zf, DFF, 0, Lezw, DIN, 0, Lezb, ub2, DIN, 0, BLT, DIN, DFF, 1, 0, 0);

        run_ssm(xx,  DFF, Lxpw,  Ldtw,  Ldtb,  LAlog,  Lqiw, Lqib, Lkiw, Lkib, true,
                ycat_in, 1024);
        run_ssm(zb,  DFF, Lxpw,  Ldtw,  Ldtb,  LAlog,  Lqiw, Lqib, Lkiw, Lkib, true,
                ycat_in + DFF, 1024);
        run_ssm(ub1, DIN, Lxpew, Ldtew, Ldteb, LAloge, Lqew, Lqeb, Lkew, Lkeb, false,
                ycat_ex, 512);
        run_ssm(ub2, DIN, Lxpew, Ldtew, Ldteb, LAloge, Lqew, Lqeb, Lkew, Lkeb, false,
                ycat_ex + DIN, 512);

        gemm(ycat_in, 1024, 0, Loiw, DFF, 0, Loib, out_inner, DFF, 0, BLT, DFF, 1024, 1, 0, 0);
        gemm(ycat_ex, 512, 0, Loew, DIN, 0, Loeb, out_exit, DIN, 0, BLT, DIN, 512, 1, 0, 0);
        {
            long total = (long)BLT * DIN;
            flip_l<<<dim3((unsigned)((total + 255) / 256)), eb, 0, stream>>>(out_exit, flipb, DIN);
        }
        gemm(flipb, DIN, 0, Lupw, DFF, 0, Lupb, upbuf, DFF, 0, BLT, DFF, DIN, 1, 0, 0);
        {
            long total = (long)BLT * 1024;
            cat_add<<<dim3((unsigned)((total + 255) / 256)), eb, 0, stream>>>(hs0, out_inner, upbuf, cat2);
        }
        gemm(cat2, 1024, 0, Loutw, DM, 0, Loutb, xbuf, DM, 0, BLT, DM, 1024, 1, 0, 0);
    }

    {
        long total = (long)BLT * DM;
        silu_out<<<dim3((unsigned)((total + 255) / 256)), eb, 0, stream>>>(
            xbuf, (float*)d_out, total);
    }
}
